// Attention_20753281975055
// MI455X (gfx1250) — compile-verified, matrix operand order checked by reading, not run on hardware
//
#include <hip/hip_runtime.h>
#include <stdint.h>

// ---------------------------------------------------------------------------
// Types for CDNA5 WMMA
// ---------------------------------------------------------------------------
typedef __attribute__((ext_vector_type(16))) __bf16 vbf16;
typedef __attribute__((ext_vector_type(8)))  float  vf8;

static __device__ __forceinline__ unsigned short f2bf(float f) {
    union { float f; unsigned int u; } c; c.f = f;
    unsigned int u = c.u;
    unsigned int r = (u + 0x7FFFu + ((u >> 16) & 1u)) >> 16;  // RNE
    return (unsigned short)r;
}

union Frag32B { vbf16 v; uint4 q[2]; };

// Load a 16-element bf16 fragment as two 16B chunks (A/B per-lane layout).
static __device__ __forceinline__ vbf16 ld_frag16(const unsigned short* c0,
                                                  const unsigned short* c1) {
    Frag32B f;
    f.q[0] = *(const uint4*)c0;
    f.q[1] = *(const uint4*)c1;
    return f.v;
}

// Async 16B global -> LDS copy (GLOBAL_LOAD_ASYNC_TO_LDS_B128, ASYNCcnt).
// Inline asm form (portable across toolchains; avoids builtin pointer-AS
// signature mismatches). LDS address = low 32 bits of the generic pointer
// (ISA: generic->LDS addresses truncate to addr[31:0]).
static __device__ __forceinline__ void cp16_async(const unsigned short* g,
                                                  unsigned short* l) {
    unsigned int la = (unsigned int)(unsigned long long)l;
    unsigned long long ga = (unsigned long long)g;
    asm volatile("global_load_async_to_lds_b128 %0, %1, off"
                 :: "v"(la), "v"(ga) : "memory");
}

static __device__ __forceinline__ void wait_async0() {
#if __has_builtin(__builtin_amdgcn_s_wait_asynccnt)
    __builtin_amdgcn_s_wait_asynccnt(0);
#else
    asm volatile("s_wait_asynccnt 0x0" ::: "memory");
#endif
}

static __device__ __forceinline__ float block_sum_256(float v, float* sm) {
    int t = threadIdx.x;
    sm[t] = v;
    __syncthreads();
    for (int off = 128; off > 0; off >>= 1) {
        if (t < off) sm[t] += sm[t + off];
        __syncthreads();
    }
    float r = sm[0];
    __syncthreads();
    return r;
}

// ---------------------------------------------------------------------------
// Problem constants
// ---------------------------------------------------------------------------
#define BB 4
#define TT 1024
#define CC 1024
#define HH 16
#define DD 64

// ---------------------------------------------------------------------------
// Kernel 1: LayerNorm(x) -> bf16 rows [B*T][C]
// ---------------------------------------------------------------------------
__global__ __launch_bounds__(256) void ln_x_kernel(
    const float* __restrict__ x, const float* __restrict__ g,
    const float* __restrict__ bta, unsigned short* __restrict__ out) {
    __shared__ float red[256];
    int row = blockIdx.x;
    int t = threadIdx.x;
    const float* xr = x + (size_t)row * CC;
    float4 xv = ((const float4*)xr)[t];
    float s = xv.x + xv.y + xv.z + xv.w;
    float mu = block_sum_256(s, red) * (1.0f / (float)CC);
    float dx = xv.x - mu, dy = xv.y - mu, dz = xv.z - mu, dw = xv.w - mu;
    float var = block_sum_256(dx * dx + dy * dy + dz * dz + dw * dw, red) *
                (1.0f / (float)CC);
    float rs = rsqrtf(var + 1e-5f);
    float4 gv = ((const float4*)g)[t];
    float4 bv = ((const float4*)bta)[t];
    unsigned short* o = out + (size_t)row * CC + t * 4;
    o[0] = f2bf(dx * rs * gv.x + bv.x);
    o[1] = f2bf(dy * rs * gv.y + bv.y);
    o[2] = f2bf(dz * rs * gv.z + bv.z);
    o[3] = f2bf(dw * rs * gv.w + bv.w);
}

// ---------------------------------------------------------------------------
// Kernel 2: transpose + convert fp32 W[K][N] -> bf16 Wt[N][K]
// ---------------------------------------------------------------------------
__global__ __launch_bounds__(256) void wtrans_kernel(
    const float* __restrict__ W, unsigned short* __restrict__ Wt, int K, int N) {
    int id = blockIdx.x * 256 + threadIdx.x;
    int n = id % N;
    int k = id / N;
    Wt[(size_t)n * K + k] = f2bf(W[(size_t)k * N + n]);
}

// ---------------------------------------------------------------------------
// Kernel 3: GEMM  C[M][N] = A[M][K](bf16) * Wt[N][K](bf16)^T + bias, fp32 out
// Block tile 128x128x32, 8 waves, each wave 64x32 (4x2 WMMA tiles).
// Double-buffered LDS staging via GLOBAL_LOAD_ASYNC_TO_LDS_B128 so the DMA
// for tile k+1 overlaps the WMMAs on tile k (ASYNCcnt synchronization).
// ---------------------------------------------------------------------------
#define GBM 128
#define GBN 128
#define GBK 32

__global__ __launch_bounds__(256) void gemm_bf16_f32(
    const unsigned short* __restrict__ A,
    const unsigned short* __restrict__ Wt,
    const float* __restrict__ bias,
    float* __restrict__ Cout, int M, int N, int K) {
    __shared__ __align__(16) unsigned short As[2][GBM * GBK];
    __shared__ __align__(16) unsigned short Bs[2][GBN * GBK];
    const int tid  = threadIdx.x;
    const int lane = tid & 31;
    const int wave = tid >> 5;
    const int wm = wave >> 2;             // 0..1 (M)
    const int wn = wave & 3;              // 0..3 (N)
    const int lr = lane & 15;
    const int lg = lane >> 4;
    const int kb = lg * 8;
    const int bM = blockIdx.y * GBM;
    const int bN = blockIdx.x * GBN;

    // Per-thread staging: two 16B chunks per matrix (rows sr and sr+64).
    const int sr = tid >> 2;              // 0..63
    const int so = (tid & 3) * 8;         // bf16 element offset in K-slab
    const unsigned short* gA0 = A  + (size_t)(bM + sr)      * K + so;
    const unsigned short* gA1 = A  + (size_t)(bM + sr + 64) * K + so;
    const unsigned short* gB0 = Wt + (size_t)(bN + sr)      * K + so;
    const unsigned short* gB1 = Wt + (size_t)(bN + sr + 64) * K + so;

    vf8 acc[4][2];
    vf8 z = {0.f, 0.f, 0.f, 0.f, 0.f, 0.f, 0.f, 0.f};
#pragma unroll
    for (int i = 0; i < 4; i++)
#pragma unroll
        for (int j = 0; j < 2; j++) acc[i][j] = z;

    const int nt = K / GBK;
    // Prologue: stage tile 0 into buffer 0.
    cp16_async(gA0, &As[0][sr * GBK + so]);
    cp16_async(gA1, &As[0][(sr + 64) * GBK + so]);
    cp16_async(gB0, &Bs[0][sr * GBK + so]);
    cp16_async(gB1, &Bs[0][(sr + 64) * GBK + so]);

    for (int kt = 0; kt < nt; kt++) {
        wait_async0();        // my chunks of tile kt have landed in LDS
        __syncthreads();      // everyone's chunks visible; prev reads done
        if (kt + 1 < nt) {
            const int nb = (kt + 1) & 1;
            const int ko = (kt + 1) * GBK;
            cp16_async(gA0 + ko, &As[nb][sr * GBK + so]);
            cp16_async(gA1 + ko, &As[nb][(sr + 64) * GBK + so]);
            cp16_async(gB0 + ko, &Bs[nb][sr * GBK + so]);
            cp16_async(gB1 + ko, &Bs[nb][(sr + 64) * GBK + so]);
        }
        const int buf = kt & 1;
        vbf16 af[4];
#pragma unroll
        for (int i = 0; i < 4; i++) {
            const unsigned short* p = &As[buf][(wm * 64 + i * 16 + lr) * GBK + kb];
            af[i] = ld_frag16(p, p + 16);
        }
        vbf16 bfr[2];
#pragma unroll
        for (int j = 0; j < 2; j++) {
            const unsigned short* p = &Bs[buf][(wn * 32 + j * 16 + lr) * GBK + kb];
            bfr[j] = ld_frag16(p, p + 16);
        }
#pragma unroll
        for (int i = 0; i < 4; i++)
#pragma unroll
            for (int j = 0; j < 2; j++)
                acc[i][j] = __builtin_amdgcn_wmma_f32_16x16x32_bf16(
                    false, af[i], false, bfr[j], (short)0, acc[i][j], false, false);
    }

#pragma unroll
    for (int i = 0; i < 4; i++)
#pragma unroll
        for (int j = 0; j < 2; j++) {
            int col = bN + wn * 32 + j * 16 + lr;
            float bv = bias[col];
#pragma unroll
            for (int r = 0; r < 8; r++) {
                int row = bM + wm * 64 + i * 16 + r + 8 * lg;
                Cout[(size_t)row * N + col] = acc[i][j][r] + bv;
            }
        }
}

// ---------------------------------------------------------------------------
// Kernel 4: Q/K LayerNorm over C + head split.
//   qh,kh: bf16 [BH][T][D]   vt: bf16 [BH][D][T] (transposed for P*V B-frags)
// ---------------------------------------------------------------------------
__global__ __launch_bounds__(256) void qkv_post_kernel(
    const float* __restrict__ qkv,
    const float* __restrict__ qg, const float* __restrict__ qb,
    const float* __restrict__ kg, const float* __restrict__ kbi,
    unsigned short* __restrict__ qh, unsigned short* __restrict__ kh,
    unsigned short* __restrict__ vt) {
    __shared__ float red[256];
    int row = blockIdx.x;          // b*T + t
    int b = row >> 10;
    int t = row & 1023;
    int tid = threadIdx.x;
    const float* qr = qkv + (size_t)row * 3 * CC;
    const float* kr = qr + CC;
    const float* vr = qr + 2 * CC;
    float4 qv = ((const float4*)qr)[tid];
    float4 kv = ((const float4*)kr)[tid];
    float4 vv = ((const float4*)vr)[tid];

    float qmu = block_sum_256(qv.x + qv.y + qv.z + qv.w, red) * (1.0f / CC);
    float qdx = qv.x - qmu, qdy = qv.y - qmu, qdz = qv.z - qmu, qdw = qv.w - qmu;
    float qvar = block_sum_256(qdx * qdx + qdy * qdy + qdz * qdz + qdw * qdw, red) *
                 (1.0f / CC);
    float qrs = rsqrtf(qvar + 1e-5f);

    float kmu = block_sum_256(kv.x + kv.y + kv.z + kv.w, red) * (1.0f / CC);
    float kdx = kv.x - kmu, kdy = kv.y - kmu, kdz = kv.z - kmu, kdw = kv.w - kmu;
    float kvar = block_sum_256(kdx * kdx + kdy * kdy + kdz * kdz + kdw * kdw, red) *
                 (1.0f / CC);
    float krs = rsqrtf(kvar + 1e-5f);

    int c0 = tid * 4;
    int h  = c0 >> 6;
    int d0 = c0 & 63;
    float4 gq = ((const float4*)qg)[tid];
    float4 bq = ((const float4*)qb)[tid];
    float4 gk = ((const float4*)kg)[tid];
    float4 bk = ((const float4*)kbi)[tid];

    size_t hb = ((size_t)(b * HH + h) * TT + t) * DD + d0;
    qh[hb + 0] = f2bf(qdx * qrs * gq.x + bq.x);
    qh[hb + 1] = f2bf(qdy * qrs * gq.y + bq.y);
    qh[hb + 2] = f2bf(qdz * qrs * gq.z + bq.z);
    qh[hb + 3] = f2bf(qdw * qrs * gq.w + bq.w);
    kh[hb + 0] = f2bf(kdx * krs * gk.x + bk.x);
    kh[hb + 1] = f2bf(kdy * krs * gk.y + bk.y);
    kh[hb + 2] = f2bf(kdz * krs * gk.z + bk.z);
    kh[hb + 3] = f2bf(kdw * krs * gk.w + bk.w);

    size_t vb = (size_t)(b * HH + h) * DD;
    vt[(vb + d0 + 0) * TT + t] = f2bf(vv.x);
    vt[(vb + d0 + 1) * TT + t] = f2bf(vv.y);
    vt[(vb + d0 + 2) * TT + t] = f2bf(vv.z);
    vt[(vb + d0 + 3) * TT + t] = f2bf(vv.w);
}

// ---------------------------------------------------------------------------
// Kernel 5: flash attention. One wave handles 16 query rows; 32-key chunks.
//   S: 2x wmma (K-dim = D = 64)       O += P*V: 4x wmma (N over D)
// Online softmax with __shfl_xor over the 16-lane groups matching C/D layout.
// ---------------------------------------------------------------------------
__global__ __launch_bounds__(128) void attn_kernel(
    const unsigned short* __restrict__ qh,
    const unsigned short* __restrict__ kh,
    const unsigned short* __restrict__ vt,
    unsigned short* __restrict__ outA) {
    __shared__ __align__(16) unsigned short plds[4 * 16 * 32];
    int bh   = blockIdx.x >> 4;          // T/64 = 16 tiles per (b,h)
    int tile = blockIdx.x & 15;
    int wave = threadIdx.x >> 5;
    int lane = threadIdx.x & 31;
    int lr = lane & 15, lg = lane >> 4;
    int kb = lg * 8;
    int t0 = tile * 64 + wave * 16;
    const unsigned short* Q  = qh + (size_t)bh * TT * DD;
    const unsigned short* Kp = kh + (size_t)bh * TT * DD;
    const unsigned short* Vp = vt + (size_t)bh * DD * TT;
    unsigned short* pl = &plds[wave * 512];

    vbf16 qa[2];
#pragma unroll
    for (int c = 0; c < 2; c++) {
        const unsigned short* p = Q + (size_t)(t0 + lr) * DD + c * 32 + kb;
        qa[c] = ld_frag16(p, p + 16);
    }
    vf8 z = {0.f, 0.f, 0.f, 0.f, 0.f, 0.f, 0.f, 0.f};
    vf8 o[4];
#pragma unroll
    for (int n = 0; n < 4; n++) o[n] = z;
    float m[8], l[8];
#pragma unroll
    for (int r = 0; r < 8; r++) { m[r] = -1e30f; l[r] = 0.f; }

    for (int k0 = 0; k0 < TT; k0 += 32) {
        vf8 s[2];
#pragma unroll
        for (int j = 0; j < 2; j++) {
            const unsigned short* pb0 = Kp + (size_t)(k0 + j * 16 + lr) * DD + kb;
            vbf16 b0 = ld_frag16(pb0, pb0 + 16);
            vbf16 b1 = ld_frag16(pb0 + 32, pb0 + 48);
            s[j] = __builtin_amdgcn_wmma_f32_16x16x32_bf16(
                false, qa[0], false, b0, (short)0, z, false, false);
            s[j] = __builtin_amdgcn_wmma_f32_16x16x32_bf16(
                false, qa[1], false, b1, (short)0, s[j], false, false);
        }
#pragma unroll
        for (int j = 0; j < 2; j++)
#pragma unroll
            for (int r = 0; r < 8; r++) s[j][r] *= 0.125f;   // D^-0.5

        float corr[8];
#pragma unroll
        for (int r = 0; r < 8; r++) {
            float v = fmaxf(s[0][r], s[1][r]);
            v = fmaxf(v, __shfl_xor(v, 1, 16));
            v = fmaxf(v, __shfl_xor(v, 2, 16));
            v = fmaxf(v, __shfl_xor(v, 4, 16));
            v = fmaxf(v, __shfl_xor(v, 8, 16));
            float nm = fmaxf(m[r], v);
            corr[r] = __expf(m[r] - nm);
            m[r] = nm;
        }
#pragma unroll
        for (int j = 0; j < 2; j++)
#pragma unroll
            for (int r = 0; r < 8; r++) s[j][r] = __expf(s[j][r] - m[r]);
#pragma unroll
        for (int r = 0; r < 8; r++) {
            float rs = s[0][r] + s[1][r];
            rs += __shfl_xor(rs, 1, 16);
            rs += __shfl_xor(rs, 2, 16);
            rs += __shfl_xor(rs, 4, 16);
            rs += __shfl_xor(rs, 8, 16);
            l[r] = l[r] * corr[r] + rs;
        }
#pragma unroll
        for (int n = 0; n < 4; n++)
#pragma unroll
            for (int r = 0; r < 8; r++) o[n][r] *= corr[r];

        // C/D layout -> A layout via LDS round-trip (bf16 probs)
#pragma unroll
        for (int j = 0; j < 2; j++)
#pragma unroll
            for (int r = 0; r < 8; r++)
                pl[(r + 8 * lg) * 32 + j * 16 + lr] = f2bf(s[j][r]);
        asm volatile("s_wait_dscnt 0x0" ::: "memory");
        vbf16 pa;
        {
            const unsigned short* pp = &pl[lr * 32 + kb];
            pa = ld_frag16(pp, pp + 16);
        }
#pragma unroll
        for (int n = 0; n < 4; n++) {
            const unsigned short* pv = Vp + (size_t)(n * 16 + lr) * TT + k0 + kb;
            vbf16 vb = ld_frag16(pv, pv + 16);
            o[n] = __builtin_amdgcn_wmma_f32_16x16x32_bf16(
                false, pa, false, vb, (short)0, o[n], false, false);
        }
    }

    int b = bh >> 4, h = bh & 15;
#pragma unroll
    for (int n = 0; n < 4; n++)
#pragma unroll
        for (int r = 0; r < 8; r++) {
            int row = b * TT + t0 + r + 8 * lg;
            int col = h * DD + n * 16 + lr;
            outA[(size_t)row * CC + col] = f2bf(o[n][r] * (1.0f / l[r]));
        }
}

// ---------------------------------------------------------------------------
// Host-side launch
// ---------------------------------------------------------------------------
extern "C" void kernel_launch(void* const* d_in, const int* in_sizes, int n_in,
                              void* d_out, int out_size, void* d_ws, size_t ws_size,
                              hipStream_t stream) {
    const float* x      = (const float*)d_in[0];
    const float* norm_g = (const float*)d_in[1];
    const float* norm_b = (const float*)d_in[2];
    const float* qkv_w  = (const float*)d_in[3];
    const float* qkv_b  = (const float*)d_in[4];
    const float* qln_g  = (const float*)d_in[5];
    const float* qln_b  = (const float*)d_in[6];
    const float* kln_g  = (const float*)d_in[7];
    const float* kln_b  = (const float*)d_in[8];
    const float* proj_w = (const float*)d_in[9];
    const float* proj_b = (const float*)d_in[10];
    float* out = (float*)d_out;

    char* ws = (char*)d_ws;
    constexpr size_t NTOK = (size_t)BB * TT;                 // 4096
    size_t off = 0;
    unsigned short* xn    = (unsigned short*)(ws + off); off += NTOK * CC * 2;
    unsigned short* WqkvT = (unsigned short*)(ws + off); off += (size_t)3 * CC * CC * 2;
    unsigned short* WpT   = (unsigned short*)(ws + off); off += (size_t)CC * CC * 2;
    float*          qkvf  = (float*)(ws + off);          off += NTOK * 3 * CC * 4;
    unsigned short* qhb   = (unsigned short*)(ws + off); off += NTOK * CC * 2;
    unsigned short* khb   = (unsigned short*)(ws + off); off += NTOK * CC * 2;
    unsigned short* vtb   = (unsigned short*)(ws + off); off += NTOK * CC * 2;
    unsigned short* attnA = (unsigned short*)(ws + off); off += NTOK * CC * 2;
    (void)ws_size; (void)in_sizes; (void)n_in; (void)out_size;

    // 1. LN(x) -> bf16
    ln_x_kernel<<<NTOK, 256, 0, stream>>>(x, norm_g, norm_b, xn);

    // 2. weight transpose/convert
    wtrans_kernel<<<(CC * 3 * CC) / 256, 256, 0, stream>>>(qkv_w, WqkvT, CC, 3 * CC);
    wtrans_kernel<<<(CC * CC) / 256, 256, 0, stream>>>(proj_w, WpT, CC, CC);

    // 3. QKV GEMM: [4096,1024] x [1024,3072]
    gemm_bf16_f32<<<dim3(3 * CC / GBN, NTOK / GBM), 256, 0, stream>>>(
        xn, WqkvT, qkv_b, qkvf, (int)NTOK, 3 * CC, CC);

    // 4. Q/K LN + head split
    qkv_post_kernel<<<NTOK, 256, 0, stream>>>(qkvf, qln_g, qln_b, kln_g, kln_b,
                                              qhb, khb, vtb);

    // 5. attention: B*H*(T/64) blocks of 4 waves
    attn_kernel<<<BB * HH * (TT / 64), 128, 0, stream>>>(qhb, khb, vtb, attnA);

    // 6. output projection: [4096,1024] x [1024,1024] -> fp32 d_out
    gemm_bf16_f32<<<dim3(CC / GBN, NTOK / GBM), 256, 0, stream>>>(
        attnA, WpT, proj_b, out, (int)NTOK, CC, CC);
}
